// DPRNN_GRU_54090818126579
// MI455X (gfx1250) — matrix operand order repaired and checked by reading, not compiled, not run on hardware
//
#include <hip/hip_runtime.h>

// ---------------------------------------------------------------------------
// DPRNN-GRU for MI455X (gfx1250, wave32, WMMA).
// All GEMMs use v_wmma_f32_16x16x32_bf16 (bf16 in, f32 accumulate).
// ---------------------------------------------------------------------------

typedef unsigned short u16;
typedef __attribute__((ext_vector_type(16))) __bf16 v16bf;
typedef __attribute__((ext_vector_type(8)))  float  v8f;

union FragU {
    v16bf v;
    u16   s[16];
    uint4 q[2];
};

__device__ __forceinline__ v8f wmma_bf16(const FragU& a, const FragU& b, v8f c) {
    // D = A(16x32 bf16) * B(32x16 bf16) + C(16x16 f32)
    return __builtin_amdgcn_wmma_f32_16x16x32_bf16(
        false, a.v, false, b.v, (short)0, c, false, false);
}

__device__ __forceinline__ u16 f2bf(float x) {
    unsigned u = __float_as_uint(x);
    u += 0x7FFFu + ((u >> 16) & 1u);          // round-to-nearest-even
    return (u16)(u >> 16);
}

// Fast gate nonlinearities: v_exp_f32 + v_rcp_f32, no IEEE-div expansion.
__device__ __forceinline__ float fast_sigmoid(float x) {
    return __builtin_amdgcn_rcpf(1.0f + __expf(-x));
}
__device__ __forceinline__ float fast_tanh(float x) {
    // tanh(x) = 1 - 2/(e^{2x}+1); saturates correctly at +/-inf exp.
    return 1.0f - 2.0f * __builtin_amdgcn_rcpf(__expf(2.0f * x) + 1.0f);
}

#define BB   16
#define CC   64
#define TT   400
#define FF   64
#define HH   64
#define GG   8
#define NSEQ (BB * TT)           // 6400 intra sequences
#define NPE  (GG * BB * (FF/GG)) // 1024 PE sequences

// ---------------------------------------------------------------------------
// K0: fp32 -> bf16 weight conversion (row-major preserved)
// ---------------------------------------------------------------------------
__global__ void cvt_bf16_kernel(const float* __restrict__ src,
                                u16* __restrict__ dst, int n) {
    int i = blockIdx.x * 256 + threadIdx.x;
    if (i < n) dst[i] = f2bf(src[i]);
}

// ---------------------------------------------------------------------------
// K1: x[b][c][t][f] -> xt bf16 [n=b*400+t][f][c]   (intra GRU input layout)
// ---------------------------------------------------------------------------
__global__ __launch_bounds__(256) void xpose_kernel(const float* __restrict__ x,
                                                    u16* __restrict__ xt) {
    int n = blockIdx.x;
    int b = n / TT, t = n % TT;
    __shared__ float sx[64 * 65];
    const float* xb = x + ((size_t)b * CC * TT + t) * FF;   // + c*TT*FF + f
    #pragma unroll
    for (int i = 0; i < 16; ++i) {
        int idx = i * 256 + threadIdx.x;
        int c = idx >> 6, f = idx & 63;
        sx[c * 65 + f] = xb[(size_t)c * (TT * FF) + f];
    }
    __syncthreads();
    u16* dst = xt + (size_t)n * (FF * CC);
    #pragma unroll
    for (int i = 0; i < 16; ++i) {
        int idx = i * 256 + threadIdx.x;
        int f = idx >> 6, c = idx & 63;
        dst[f * 64 + c] = f2bf(sx[c * 65 + f]);
    }
}

// ---------------------------------------------------------------------------
// K2/K4: GRU scan.  16 sequences per block, 4 waves (one 16-col window each).
// seq: bf16 [Nrows][L][64].  Weights bf16 row-major [3H][H] per group.
// out: bf16 [Nrows][L][64]  (hidden states per step).
// Hidden state double-buffered in LDS -> one barrier per recurrence step.
// ---------------------------------------------------------------------------
__global__ __launch_bounds__(128) void gru_scan_kernel(
    const u16* __restrict__ seq, int L,
    const u16* __restrict__ wih_base, const u16* __restrict__ whh_base,
    const float* __restrict__ bih_base, const float* __restrict__ bhh_base,
    u16* __restrict__ out, int reverse, int blocks_per_group)
{
    const int tid  = threadIdx.x;
    const int lane = tid & 31;
    const int w    = tid >> 5;         // wave 0..3 -> column window
    const int half = lane >> 4;
    const int m    = lane & 15;
    const int bx   = blockIdx.x;
    const int g    = bx / blocks_per_group;
    const int n0   = bx * 16;

    const u16*   wih = wih_base + (size_t)g * (3 * HH * HH);
    const u16*   whh = whh_base + (size_t)g * (3 * HH * HH);
    const float* bih = bih_base + g * (3 * HH);
    const float* bhh = bhh_base + g * (3 * HH);

    __shared__ float h_f[2][16 * 64];
    __shared__ u16   h_b[2][16 * 64];
    for (int i = tid; i < 1024; i += 128) {
        h_f[0][i] = 0.0f; h_b[0][i] = 0;
    }
    __syncthreads();

    const int col = w * 16 + m;        // hidden index for this lane's C column

    const float bir = bih[col],        biz = bih[64 + col],  bin_ = bih[128 + col];
    const float bhr = bhh[col],        bhz = bhh[64 + col],  bhn  = bhh[128 + col];

    // Resident recurrent-weight B-fragments: [gate r/z/n][k-chunk 0/1]
    FragU Bi[3][2], Bh[3][2];
    #pragma unroll
    for (int gate = 0; gate < 3; ++gate) {
        const int jrow = gate * 64 + col;
        #pragma unroll
        for (int kc = 0; kc < 2; ++kc) {
            const u16* p = wih + jrow * 64 + kc * 32 + half * 8;
            Bi[gate][kc].q[0] = *(const uint4*)(p);
            Bi[gate][kc].q[1] = *(const uint4*)(p + 16);
            const u16* q = whh + jrow * 64 + kc * 32 + half * 8;
            Bh[gate][kc].q[0] = *(const uint4*)(q);
            Bh[gate][kc].q[1] = *(const uint4*)(q + 16);
        }
    }

    const size_t Ls   = (size_t)L * 64;
    const u16*   arow = seq + (size_t)(n0 + m) * Ls;   // A row = this lane's seq
    u16*         ob   = out + (size_t)n0 * Ls;
    const int    ao   = half * 8;                      // A-fragment K sub-offset

    for (int s = 0; s < L; ++s) {
        const int cur = s & 1, nxt = cur ^ 1;
        const int f_io = reverse ? (L - 1 - s) : s;
        const u16* xr  = arow + (size_t)f_io * 64;

        FragU ax0, ax1, ah0, ah1;
        ax0.q[0] = *(const uint4*)(xr + ao);
        ax0.q[1] = *(const uint4*)(xr + ao + 16);
        ax1.q[0] = *(const uint4*)(xr + 32 + ao);
        ax1.q[1] = *(const uint4*)(xr + 32 + ao + 16);
        const u16* hb = h_b[cur] + m * 64;
        ah0.q[0] = *(const uint4*)(hb + ao);
        ah0.q[1] = *(const uint4*)(hb + ao + 16);
        ah1.q[0] = *(const uint4*)(hb + 32 + ao);
        ah1.q[1] = *(const uint4*)(hb + 32 + ao + 16);

        if (s + 1 < L) {  // hide next step's strided activation fetch
            const int fn = reverse ? (L - 2 - s) : (s + 1);
            __builtin_prefetch(arow + (size_t)fn * 64, 0, 3);
        }

        v8f gx[3], gh[3];
        #pragma unroll
        for (int gate = 0; gate < 3; ++gate) {
            v8f zx = {0, 0, 0, 0, 0, 0, 0, 0};
            zx = wmma_bf16(ax0, Bi[gate][0], zx);
            zx = wmma_bf16(ax1, Bi[gate][1], zx);
            gx[gate] = zx;
            v8f zh = {0, 0, 0, 0, 0, 0, 0, 0};
            zh = wmma_bf16(ah0, Bh[gate][0], zh);
            zh = wmma_bf16(ah1, Bh[gate][1], zh);
            gh[gate] = zh;
        }

        float ho[8];
        #pragma unroll
        for (int v = 0; v < 8; ++v) ho[v] = h_f[cur][(v + 8 * half) * 64 + col];

        #pragma unroll
        for (int v = 0; v < 8; ++v) {
            const float r  = fast_sigmoid(gx[0][v] + bir + gh[0][v] + bhr);
            const float z_ = fast_sigmoid(gx[1][v] + biz + gh[1][v] + bhz);
            const float nn = fast_tanh(gx[2][v] + bin_ + r * (gh[2][v] + bhn));
            const float hv = (1.0f - z_) * nn + z_ * ho[v];
            const int row  = v + 8 * half;
            h_f[nxt][row * 64 + col] = hv;
            const u16 hb16 = f2bf(hv);
            h_b[nxt][row * 64 + col] = hb16;
            ob[(size_t)row * Ls + (size_t)f_io * 64 + col] = hb16;
        }
        __syncthreads();   // next-step buffer complete; prev reads also done
    }
}

// ---------------------------------------------------------------------------
// K3: intra = LN2(concat[fwd,bwd] @ fcW^T + b); write intra f32 and
//     yt = bf16(xp + intra) in PE layout [g*128 + b*8 + fgi][t][c].
//     One block per (b,t): GEMM M=64(f), N=64(c), K=128; 8 waves, 2 tiles each.
// ---------------------------------------------------------------------------
__global__ __launch_bounds__(256) void intra_fc_ln_kernel(
    const u16* __restrict__ fwd, const u16* __restrict__ bwd,
    const u16* __restrict__ fcw,        // bf16 [64][128]
    const float* __restrict__ fcb,
    const float* __restrict__ lng, const float* __restrict__ lnb,
    const float* __restrict__ x,
    float* __restrict__ intra, u16* __restrict__ yt)
{
    const int n = blockIdx.x;
    const int b = n / TT, t = n % TT;
    const int tid = threadIdx.x, lane = tid & 31, w = tid >> 5;
    const int half = lane >> 4, m = lane & 15;

    __shared__ float s[64 * 65];
    __shared__ float red[256];

    const u16* fr = fwd + (size_t)n * 4096;
    const u16* br = bwd + (size_t)n * 4096;

    #pragma unroll
    for (int ti = 0; ti < 2; ++ti) {
        const int tile = w * 2 + ti;
        const int mt = tile >> 2, nt = tile & 3;
        const int co = nt * 16 + m;
        v8f acc;
        const float bb = fcb[co];
        #pragma unroll
        for (int v = 0; v < 8; ++v) acc[v] = bb;
        const u16* arow_f = fr + (mt * 16 + m) * 64;
        const u16* arow_b = br + (mt * 16 + m) * 64;
        #pragma unroll
        for (int kc = 0; kc < 4; ++kc) {
            FragU a, bw;
            const u16* ap = (kc < 2 ? arow_f : arow_b) + (kc & 1) * 32 + half * 8;
            a.q[0] = *(const uint4*)(ap);
            a.q[1] = *(const uint4*)(ap + 16);
            const u16* bp = fcw + co * 128 + kc * 32 + half * 8;
            bw.q[0] = *(const uint4*)(bp);
            bw.q[1] = *(const uint4*)(bp + 16);
            acc = wmma_bf16(a, bw, acc);
        }
        #pragma unroll
        for (int v = 0; v < 8; ++v)
            s[(mt * 16 + v + 8 * half) * 65 + co] = acc[v];
    }
    __syncthreads();

    // LayerNorm statistics over all 4096 elements
    float s1 = 0.f, s2 = 0.f;
    #pragma unroll
    for (int i = 0; i < 16; ++i) {
        const int idx = tid + 256 * i;
        const float v = s[(idx >> 6) * 65 + (idx & 63)];
        s1 += v; s2 += v * v;
    }
    red[tid] = s1; __syncthreads();
    for (int off = 128; off > 0; off >>= 1) {
        if (tid < off) red[tid] += red[tid + off];
        __syncthreads();
    }
    const float mean = red[0] * (1.0f / 4096.0f);
    __syncthreads();
    red[tid] = s2; __syncthreads();
    for (int off = 128; off > 0; off >>= 1) {
        if (tid < off) red[tid] += red[tid + off];
        __syncthreads();
    }
    const float var  = red[0] * (1.0f / 4096.0f) - mean * mean;
    const float rstd = rsqrtf(var + 1e-8f);

    float* ip = intra + (size_t)n * 4096;
    const float* xb = x + ((size_t)b * CC * TT + t) * FF;
    #pragma unroll
    for (int i = 0; i < 16; ++i) {
        const int idx = tid + 256 * i;
        const int f = idx >> 6, c = idx & 63;
        const float nv = (s[f * 65 + c] - mean) * rstd * lng[f * 64 + c] + lnb[f * 64 + c];
        ip[f * 64 + c] = nv;
        const float xp = xb[(size_t)c * (TT * FF) + f];
        const size_t yrow = (size_t)((f >> 3) * 128 + b * 8 + (f & 7));
        yt[yrow * (TT * 64) + (size_t)t * 64 + c] = f2bf(nv + xp);
    }
}

// ---------------------------------------------------------------------------
// K5: PE FC.  rnn bf16 flat [409600][64] @ pe_fc_W[g]^T -> fc f32 [409600][64]
//     64 rows per block (rows per group = 51200, divisible by 64).
// ---------------------------------------------------------------------------
__global__ __launch_bounds__(256) void pe_fc_kernel(
    const u16* __restrict__ rnn, const u16* __restrict__ fcw /* bf16 [8][64][64] */,
    const float* __restrict__ fcb /* [8][64] */, float* __restrict__ fc)
{
    const int bx = blockIdx.x;
    const int g  = bx / 800;
    const size_t R0 = (size_t)bx * 64;
    const int tid = threadIdx.x, lane = tid & 31, w = tid >> 5;
    const int half = lane >> 4, m = lane & 15;

    #pragma unroll
    for (int ti = 0; ti < 2; ++ti) {
        const int tile = w * 2 + ti;
        const int mt = tile >> 2, nt = tile & 3;
        const int co = nt * 16 + m;
        v8f acc;
        const float bb = fcb[g * 64 + co];
        #pragma unroll
        for (int v = 0; v < 8; ++v) acc[v] = bb;
        const u16* arow = rnn + (R0 + mt * 16 + m) * 64;
        const u16* wrow = fcw + (size_t)g * 4096 + co * 64;
        #pragma unroll
        for (int kc = 0; kc < 2; ++kc) {
            FragU a, bw;
            const u16* ap = arow + kc * 32 + half * 8;
            a.q[0] = *(const uint4*)(ap);
            a.q[1] = *(const uint4*)(ap + 16);
            const u16* bp = wrow + kc * 32 + half * 8;
            bw.q[0] = *(const uint4*)(bp);
            bw.q[1] = *(const uint4*)(bp + 16);
            acc = wmma_bf16(a, bw, acc);
        }
        #pragma unroll
        for (int v = 0; v < 8; ++v)
            fc[(R0 + mt * 16 + v + 8 * half) * 64 + co] = acc[v];
    }
}

// ---------------------------------------------------------------------------
// K6: inter = LN2(fc gathered to [b,t,f,c]); out[b][c][t][f] = intra + inter
// ---------------------------------------------------------------------------
__global__ __launch_bounds__(256) void final_ln_add_kernel(
    const float* __restrict__ fc, const float* __restrict__ intra,
    const float* __restrict__ lng, const float* __restrict__ lnb,
    float* __restrict__ out)
{
    const int n = blockIdx.x;
    const int b = n / TT, t = n % TT;
    const int tid = threadIdx.x;

    __shared__ float s[64 * 65];
    __shared__ float red[256];

    #pragma unroll
    for (int i = 0; i < 16; ++i) {
        const int idx = tid + 256 * i;
        const int f = idx >> 6, c = idx & 63;
        const size_t row = (size_t)((f >> 3) * 128 + b * 8 + (f & 7)) * TT + t;
        s[f * 65 + c] = fc[row * 64 + c];
    }
    __syncthreads();

    float s1 = 0.f, s2 = 0.f;
    #pragma unroll
    for (int i = 0; i < 16; ++i) {
        const int idx = tid + 256 * i;
        const float v = s[(idx >> 6) * 65 + (idx & 63)];
        s1 += v; s2 += v * v;
    }
    red[tid] = s1; __syncthreads();
    for (int off = 128; off > 0; off >>= 1) {
        if (tid < off) red[tid] += red[tid + off];
        __syncthreads();
    }
    const float mean = red[0] * (1.0f / 4096.0f);
    __syncthreads();
    red[tid] = s2; __syncthreads();
    for (int off = 128; off > 0; off >>= 1) {
        if (tid < off) red[tid] += red[tid + off];
        __syncthreads();
    }
    const float var  = red[0] * (1.0f / 4096.0f) - mean * mean;
    const float rstd = rsqrtf(var + 1e-8f);

    const float* ip = intra + (size_t)n * 4096;
    const int c  = tid >> 2;
    const int fq = tid & 3;
    float* ob = out + ((size_t)(b * CC + c) * TT + t) * FF;
    #pragma unroll
    for (int j = 0; j < 4; ++j) {
        const int f0 = fq * 16 + j * 4;
        float4 o;
        float* op = (float*)&o;
        #pragma unroll
        for (int k = 0; k < 4; ++k) {
            const int f = f0 + k;
            const float iv = (s[f * 65 + c] - mean) * rstd * lng[f * 64 + c] + lnb[f * 64 + c];
            op[k] = ip[f * 64 + c] + iv;
        }
        *(float4*)(ob + f0) = o;
    }
}

// ---------------------------------------------------------------------------
// Host-side launch
// ---------------------------------------------------------------------------
extern "C" void kernel_launch(void* const* d_in, const int* in_sizes, int n_in,
                              void* d_out, int out_size, void* d_ws, size_t ws_size,
                              hipStream_t stream) {
    (void)in_sizes; (void)n_in; (void)out_size; (void)ws_size;

    const float* x        = (const float*)d_in[0];
    const float* iWih_f   = (const float*)d_in[1];
    const float* iWhh_f   = (const float*)d_in[2];
    const float* ibih_f   = (const float*)d_in[3];
    const float* ibhh_f   = (const float*)d_in[4];
    const float* iWih_b   = (const float*)d_in[5];
    const float* iWhh_b   = (const float*)d_in[6];
    const float* ibih_b   = (const float*)d_in[7];
    const float* ibhh_b   = (const float*)d_in[8];
    const float* ifc_W    = (const float*)d_in[9];
    const float* ifc_b    = (const float*)d_in[10];
    const float* iln_g    = (const float*)d_in[11];
    const float* iln_b    = (const float*)d_in[12];
    const float* pWih     = (const float*)d_in[13];
    const float* pWhh     = (const float*)d_in[14];
    const float* pbih     = (const float*)d_in[15];
    const float* pbhh     = (const float*)d_in[16];
    const float* pfc_W    = (const float*)d_in[17];
    const float* pfc_b    = (const float*)d_in[18];
    const float* pln_g    = (const float*)d_in[19];
    const float* pln_b    = (const float*)d_in[20];

    char* ws = (char*)d_ws;

    // bf16 weight region (element offsets into u16 array)
    u16* w16 = (u16*)ws;
    const size_t WIHF = 0,       WHHF = 12288, WIHB = 24576, WHHB = 36864;
    const size_t PEWIH = 49152,  PEWHH = 147456;
    const size_t FCW  = 245760,  PEFCW = 253952;     // end 286720 u16

    const size_t SEQ_BYTES   = (size_t)NSEQ * 4096 * 2;   // 52,428,800
    const size_t INTRA_BYTES = (size_t)NSEQ * 4096 * 4;   // 104,857,600
    const size_t XT_OFF    = 1u << 20;
    const size_t FWD_OFF   = XT_OFF + SEQ_BYTES;
    const size_t BWD_OFF   = FWD_OFF + SEQ_BYTES;
    const size_t INTRA_OFF = BWD_OFF + SEQ_BYTES;
    const size_t FC_OFF    = INTRA_OFF + INTRA_BYTES;

    u16*   xt    = (u16*)(ws + XT_OFF);
    u16*   fwd   = (u16*)(ws + FWD_OFF);
    u16*   bwd   = (u16*)(ws + BWD_OFF);
    float* intra = (float*)(ws + INTRA_OFF);
    u16*   yt    = (u16*)(ws + XT_OFF);    // alias xt (dead after intra scans)
    u16*   rnn   = (u16*)(ws + FWD_OFF);   // alias fwd (dead after K3)
    float* fcbuf = (float*)(ws + FC_OFF);
    float* outp  = (float*)d_out;

    // K0: weight conversions
    cvt_bf16_kernel<<<(12288 + 255) / 256, 256, 0, stream>>>(iWih_f, w16 + WIHF, 12288);
    cvt_bf16_kernel<<<(12288 + 255) / 256, 256, 0, stream>>>(iWhh_f, w16 + WHHF, 12288);
    cvt_bf16_kernel<<<(12288 + 255) / 256, 256, 0, stream>>>(iWih_b, w16 + WIHB, 12288);
    cvt_bf16_kernel<<<(12288 + 255) / 256, 256, 0, stream>>>(iWhh_b, w16 + WHHB, 12288);
    cvt_bf16_kernel<<<(98304 + 255) / 256, 256, 0, stream>>>(pWih,   w16 + PEWIH, 98304);
    cvt_bf16_kernel<<<(98304 + 255) / 256, 256, 0, stream>>>(pWhh,   w16 + PEWHH, 98304);
    cvt_bf16_kernel<<<(8192  + 255) / 256, 256, 0, stream>>>(ifc_W,  w16 + FCW,   8192);
    cvt_bf16_kernel<<<(32768 + 255) / 256, 256, 0, stream>>>(pfc_W,  w16 + PEFCW, 32768);

    // K1: pack x -> bf16 [n][f][c]
    xpose_kernel<<<NSEQ, 256, 0, stream>>>(x, xt);

    // K2: intra bidirectional GRU along frequency (L=64)
    gru_scan_kernel<<<NSEQ / 16, 128, 0, stream>>>(
        xt, FF, w16 + WIHF, w16 + WHHF, ibih_f, ibhh_f, fwd, 0, NSEQ / 16);
    gru_scan_kernel<<<NSEQ / 16, 128, 0, stream>>>(
        xt, FF, w16 + WIHB, w16 + WHHB, ibih_b, ibhh_b, bwd, 1, NSEQ / 16);

    // K3: intra FC + LN + residual, pack PE input
    intra_fc_ln_kernel<<<NSEQ, 256, 0, stream>>>(
        fwd, bwd, w16 + FCW, ifc_b, iln_g, iln_b, x, intra, yt);

    // K4: PE grouped GRUs along time (L=400), 8 blocks per group
    gru_scan_kernel<<<NPE / 16, 128, 0, stream>>>(
        yt, TT, w16 + PEWIH, w16 + PEWHH, pbih, pbhh, rnn, 0, 8);

    // K5: PE FC
    pe_fc_kernel<<<(NPE * TT) / 64, 256, 0, stream>>>(rnn, w16 + PEFCW, pfc_b, fcbuf);

    // K6: final LN + residual + transpose to [b][c][t][f]
    final_ln_add_kernel<<<NSEQ, 256, 0, stream>>>(fcbuf, intra, pln_g, pln_b, outp);
}
